// CausalGAT_41721312313877
// MI455X (gfx1250) — matrix operand order repaired and checked by reading, not compiled, not run on hardware
//
#include <hip/hip_runtime.h>
#include <math.h>

#define NN   50000     // nodes
#define NE   800000    // edges
#define HD   128       // hidden / feature dim
#define NH   4         // heads
#define NC   10        // classes
#define NG   500       // graphs

typedef _Float16 v16h __attribute__((ext_vector_type(16)));
typedef _Float16 v8h  __attribute__((ext_vector_type(8)));
typedef float    v8f  __attribute__((ext_vector_type(8)));

union ABfrag { v16h v; v8h h[2]; };

// ---------------------------------------------------------------------------
// Fused GEMM: Out[n, :128] = act( (rs[n]*A[n,k]*cs[k]*bnmul + sh[k]) @ W + bias )
// K = M = 128 fixed. One block = 256 threads = 8 waves; each wave does one
// 16x16 tile via v_wmma_f32_16x16x32_f16, K-loop of 4.
// ---------------------------------------------------------------------------
__global__ __launch_bounds__(256) void k_gemm128(
    const float* __restrict__ A, const float* __restrict__ W,
    const float* __restrict__ bias,
    const float* __restrict__ colscale, const float* __restrict__ colshift,
    float bnmul,
    const float* __restrict__ rowscale, int rs_stride, int rs_off,
    float* __restrict__ Out, int nrows, int relu)
{
  __shared__ _Float16 sA[16 * 128];     // A tile, row-major [r][k]
  __shared__ _Float16 sW[128 * 128];    // W transposed:   [m][k]
  const int tid  = threadIdx.x;
  const int row0 = blockIdx.x * 16;

  // Stage A tile (with BN / row-scale transform) as f16
  for (int i = tid; i < 16 * 128; i += 256) {
    int r = i >> 7, k = i & 127;
    int gr = row0 + r;
    float a = 0.f;
    if (gr < nrows) {
      a = A[(long)gr * HD + k];
      if (rowscale) a *= rowscale[(long)gr * rs_stride + rs_off];
      if (colscale) a *= colscale[k] * bnmul;
      if (colshift) a += colshift[k];
    }
    sA[i] = (_Float16)a;
  }
  // Stage W (transposed) as f16
  for (int i = tid; i < 128 * 128; i += 256) {
    int k = i >> 7, m = i & 127;
    sW[m * 128 + k] = (_Float16)W[i];
  }
  __syncthreads();

  const int wave = tid >> 5;
  const int lane = tid & 31;
  const int half = lane >> 4;   // lane group 0..15 / 16..31
  const int lr   = lane & 15;
  const int col0 = wave * 16;

  v8f acc = {};
#pragma unroll
  for (int kt = 0; kt < 4; ++kt) {
    ABfrag a, b;
    // A 16x32 f16 layout: lanes 0-15 hold K[0..7],[16..23]; lanes 16-31 K[8..15],[24..31]
    const _Float16* ap = &sA[lr * 128 + kt * 32 + 8 * half];
    a.h[0] = *(const v8h*)(ap);
    a.h[1] = *(const v8h*)(ap + 16);
    // B 32x16 f16 layout: lanes 0-15 hold K[0..7],[16..23]; lanes 16-31 K[8..15],[24..31] (col = lr)
    const _Float16* bp = &sW[(col0 + lr) * 128 + kt * 32 + 8 * half];
    b.h[0] = *(const v8h*)(bp);
    b.h[1] = *(const v8h*)(bp + 16);
    acc = __builtin_amdgcn_wmma_f32_16x16x32_f16(
        false, a.v, false, b.v, (short)0, acc, false, false);
  }

  // D layout: VGPR v -> row v + 8*half, col = lr
#pragma unroll
  for (int v = 0; v < 8; ++v) {
    int gr = row0 + v + 8 * half;
    if (gr >= nrows) continue;
    int gc = col0 + lr;
    float o = acc[v];
    if (bias) o += bias[gc];
    if (relu) o = fmaxf(o, 0.f);
    Out[(long)gr * HD + gc] = o;
  }
}

// ---------------------------------------------------------------------------
// GAT helpers
// ---------------------------------------------------------------------------
__global__ void k_alpha(const float* __restrict__ T,
                        const float* __restrict__ aw_src, const float* __restrict__ aw_dst,
                        float* __restrict__ asrc, float* __restrict__ adst)
{
  int i = blockIdx.x * blockDim.x + threadIdx.x;      // n*NH + h
  if (i >= NN * NH) return;
  int n = i >> 2, h = i & 3;
  const float* hp = T + (long)n * HD + h * 32;
  const float* ws = aw_src + h * 32;
  const float* wd = aw_dst + h * 32;
  float s = 0.f, d = 0.f;
#pragma unroll 8
  for (int k = 0; k < 32; ++k) { float v = hp[k]; s += v * ws[k]; d += v * wd[k]; }
  asrc[i] = s; adst[i] = d;
}

__device__ __forceinline__ unsigned f2key(float f) {
  unsigned u = __float_as_uint(f);
  return (u & 0x80000000u) ? ~u : (u | 0x80000000u);
}
__device__ __forceinline__ float key2f(unsigned k) {
  return (k & 0x80000000u) ? __uint_as_float(k & 0x7fffffffu) : __uint_as_float(~k);
}

__global__ void k_edge_logit(const int* __restrict__ row, const int* __restrict__ col,
                             const float* __restrict__ asrc, const float* __restrict__ adst,
                             float* __restrict__ att, unsigned* __restrict__ mkey)
{
  int e = blockIdx.x * blockDim.x + threadIdx.x;
  if (e >= NE + NN) return;
  int s = e < NE ? row[e] : e - NE;
  int d = e < NE ? col[e] : e - NE;
#pragma unroll
  for (int h = 0; h < NH; ++h) {
    float a = asrc[s * NH + h] + adst[d * NH + h];
    a = a > 0.f ? a : 0.2f * a;                       // leaky_relu 0.2
    att[(long)e * NH + h] = a;
    atomicMax(&mkey[d * NH + h], f2key(a));
  }
}

__global__ void k_edge_exp(const int* __restrict__ row, const int* __restrict__ col,
                           const unsigned* __restrict__ mkey,
                           float* __restrict__ att, float* __restrict__ ssum)
{
  int e = blockIdx.x * blockDim.x + threadIdx.x;
  if (e >= NE + NN) return;
  int d = e < NE ? col[e] : e - NE;
#pragma unroll
  for (int h = 0; h < NH; ++h) {
    float m  = key2f(mkey[d * NH + h]);
    float ex = __expf(att[(long)e * NH + h] - m);
    att[(long)e * NH + h] = ex;
    atomicAdd(&ssum[d * NH + h], ex);
  }
}

__global__ void k_gat_scatter(const int* __restrict__ row, const int* __restrict__ col,
                              const float* __restrict__ att, const float* __restrict__ ssum,
                              const float* __restrict__ T, float* __restrict__ S)
{
  long long i = (long long)blockIdx.x * blockDim.x + threadIdx.x;
  if (i >= (long long)(NE + NN) * HD) return;
  int f = (int)(i & 127);
  int e = (int)(i >> 7);
  int s = e < NE ? row[e] : e - NE;
  int d = e < NE ? col[e] : e - NE;
  int h = f >> 5;
  float al = att[(long)e * NH + h] / (ssum[d * NH + h] + 1e-16f);
  atomicAdd(&S[(long)d * HD + f], al * T[(long)s * HD + f]);
}

__global__ void k_bias_relu(const float* __restrict__ S, const float* __restrict__ b,
                            float* __restrict__ X)
{
  int i = blockIdx.x * blockDim.x + threadIdx.x;
  if (i >= NN * HD) return;
  X[i] = fmaxf(S[i] + b[i & 127], 0.f);
}

// ---------------------------------------------------------------------------
// Edge / node attention MLPs (softmax over 2 outputs)
// ---------------------------------------------------------------------------
__global__ void k_edge_att(const int* __restrict__ row, const int* __restrict__ col,
                           const float* __restrict__ X, const float* __restrict__ W,
                           const float* __restrict__ b, float* __restrict__ eatt)
{
  int e = blockIdx.x * blockDim.x + threadIdx.x;
  if (e >= NE) return;
  const float* xr = X + (long)row[e] * HD;
  const float* xc = X + (long)col[e] * HD;
  float l0 = b[0], l1 = b[1];
#pragma unroll 4
  for (int k = 0; k < HD; ++k) { float v = xr[k]; l0 += v * W[k * 2]; l1 += v * W[k * 2 + 1]; }
#pragma unroll 4
  for (int k = 0; k < HD; ++k) { float v = xc[k]; l0 += v * W[(HD + k) * 2]; l1 += v * W[(HD + k) * 2 + 1]; }
  float mx = fmaxf(l0, l1);
  float e0 = __expf(l0 - mx), e1 = __expf(l1 - mx);
  float inv = 1.f / (e0 + e1);
  eatt[(long)e * 2]     = e0 * inv;
  eatt[(long)e * 2 + 1] = e1 * inv;
}

__global__ void k_node_att(const float* __restrict__ X, const float* __restrict__ W,
                           const float* __restrict__ b, float* __restrict__ natt)
{
  int n = blockIdx.x * blockDim.x + threadIdx.x;
  if (n >= NN) return;
  const float* xp = X + (long)n * HD;
  float l0 = b[0], l1 = b[1];
#pragma unroll 4
  for (int k = 0; k < HD; ++k) { float v = xp[k]; l0 += v * W[k * 2]; l1 += v * W[k * 2 + 1]; }
  float mx = fmaxf(l0, l1);
  float e0 = __expf(l0 - mx), e1 = __expf(l1 - mx);
  float inv = 1.f / (e0 + e1);
  natt[(long)n * 2]     = e0 * inv;
  natt[(long)n * 2 + 1] = e1 * inv;
}

// ---------------------------------------------------------------------------
// GCN helpers
// ---------------------------------------------------------------------------
__global__ void k_deg(const int* __restrict__ row, const float* __restrict__ eatt,
                      int sel, float* __restrict__ deg)
{
  int e = blockIdx.x * blockDim.x + threadIdx.x;
  if (e >= NE + NN) return;
  if (e < NE) atomicAdd(&deg[row[e]], eatt[(long)e * 2 + sel]);
  else        atomicAdd(&deg[e - NE], 1.0f);
}

__global__ void k_dis(float* __restrict__ deg)
{
  int n = blockIdx.x * blockDim.x + threadIdx.x;
  if (n >= NN) return;
  float d = deg[n];
  deg[n] = d > 0.f ? rsqrtf(d) : 0.f;
}

__global__ void k_gcn_scatter(const int* __restrict__ row, const int* __restrict__ col,
                              const float* __restrict__ eatt, int sel,
                              const float* __restrict__ dis,
                              const float* __restrict__ T, float* __restrict__ S)
{
  long long i = (long long)blockIdx.x * blockDim.x + threadIdx.x;
  if (i >= (long long)(NE + NN) * HD) return;
  int f = (int)(i & 127);
  int e = (int)(i >> 7);
  int r = e < NE ? row[e] : e - NE;
  int c = e < NE ? col[e] : e - NE;
  float w  = e < NE ? eatt[(long)e * 2 + sel] : 1.0f;
  float nm = dis[r] * w * dis[c];
  atomicAdd(&S[(long)c * HD + f], nm * T[(long)r * HD + f]);
}

__global__ void k_bias_relu_pool(const float* __restrict__ S, const float* __restrict__ b,
                                 const int* __restrict__ batch, float* __restrict__ pool)
{
  int i = blockIdx.x * blockDim.x + threadIdx.x;
  if (i >= NN * HD) return;
  int n = i >> 7, f = i & 127;
  float v = fmaxf(S[i] + b[f], 0.f);
  atomicAdd(&pool[(long)batch[n] * HD + f], v);
}

__global__ void k_combine(const float* __restrict__ pc, const float* __restrict__ po,
                          const int* __restrict__ perm, float* __restrict__ pco)
{
  int i = blockIdx.x * blockDim.x + threadIdx.x;
  if (i >= NG * HD) return;
  int g = i >> 7, f = i & 127;
  pco[i] = pc[(long)perm[g] * HD + f] + po[i];
}

// Readout tail: bn2 -> @W2(128x10) + b2 -> log_softmax.  One wave per graph.
__global__ void k_head(const float* __restrict__ R,
                       const float* __restrict__ bn2w, const float* __restrict__ bn2b,
                       float bnmul,
                       const float* __restrict__ W2, const float* __restrict__ b2,
                       float* __restrict__ out)
{
  int g = blockIdx.x;
  int lane = threadIdx.x;
  float acc[NC];
#pragma unroll
  for (int c = 0; c < NC; ++c) acc[c] = 0.f;
  for (int k = lane; k < HD; k += 32) {
    float v = R[(long)g * HD + k] * bn2w[k] * bnmul + bn2b[k];
#pragma unroll
    for (int c = 0; c < NC; ++c) acc[c] += v * W2[k * NC + c];
  }
#pragma unroll
  for (int off = 16; off > 0; off >>= 1) {
#pragma unroll
    for (int c = 0; c < NC; ++c) acc[c] += __shfl_down(acc[c], off);
  }
  if (lane == 0) {
    float l[NC]; float mx = -1e30f;
#pragma unroll
    for (int c = 0; c < NC; ++c) { l[c] = acc[c] + b2[c]; mx = fmaxf(mx, l[c]); }
    float s = 0.f;
#pragma unroll
    for (int c = 0; c < NC; ++c) s += __expf(l[c] - mx);
    float lse = mx + logf(s);
#pragma unroll
    for (int c = 0; c < NC; ++c) out[(long)g * NC + c] = l[c] - lse;
  }
}

// ---------------------------------------------------------------------------
extern "C" void kernel_launch(void* const* d_in, const int* in_sizes, int n_in,
                              void* d_out, int out_size, void* d_ws, size_t ws_size,
                              hipStream_t stream)
{
  (void)in_sizes; (void)n_in; (void)out_size; (void)ws_size;
  const float* x    = (const float*)d_in[0];
  const int*   ei   = (const int*)d_in[1];
  const int*   row  = ei;
  const int*   col  = ei + NE;
  const int*   batch= (const int*)d_in[2];
  const int*   perm = (const int*)d_in[3];
  const float* P[64];
  for (int i = 4; i < n_in && i < 64; ++i) P[i] = (const float*)d_in[i];
  // param indices (setup_inputs dict order, offset by 4 tensor inputs):
  // 4 bn_feat_w 5 bn_feat_b 6 conv_feat_W 7 conv_feat_b 8 edge_att_W 9 edge_att_b
  // 10 node_att_W 11 node_att_b 12 bnc_w 13 bnc_b 14 bno_w 15 bno_b
  // 16 ctx_W 17 ctx_b 18 obj_W 19 obj_b
  // 20+6i: gat{i}_bn_w,bn_b,W,a_src,a_dst,b   (i=0,1)
  // 32+8j: {c,o,co}_bn1_w,bn1_b,W1,b1,bn2_w,bn2_b,W2,b2

  const float BNM = 1.0f / sqrtf(1.0f + 1e-5f);

  // --- workspace carve ---
  char* wp = (char*)d_ws;
  auto carve = [&](size_t bytes) -> void* {
    void* p = (void*)wp; wp += (bytes + 255) & ~(size_t)255; return p;
  };
  float*    X     = (float*)   carve((size_t)NN * HD * 4);
  float*    T     = (float*)   carve((size_t)NN * HD * 4);
  float*    S     = (float*)   carve((size_t)NN * HD * 4);
  float*    att   = (float*)   carve((size_t)(NE + NN) * NH * 4);
  float*    asrc  = (float*)   carve((size_t)NN * NH * 4);
  float*    adst  = (float*)   carve((size_t)NN * NH * 4);
  unsigned* mkey  = (unsigned*)carve((size_t)NN * NH * 4);
  float*    ssum  = (float*)   carve((size_t)NN * NH * 4);
  float*    eatt  = (float*)   carve((size_t)NE * 2 * 4);
  float*    natt  = (float*)   carve((size_t)NN * 2 * 4);
  float*    deg   = (float*)   carve((size_t)NN * 4);
  float*    poolc = (float*)   carve((size_t)NG * HD * 4);
  float*    poolo = (float*)   carve((size_t)NG * HD * 4);
  float*    pco   = (float*)   carve((size_t)NG * HD * 4);
  float*    R     = (float*)   carve((size_t)NG * HD * 4);

  auto gemm = [&](const float* A, const float* W, const float* bias,
                  const float* cs, const float* sh,
                  const float* rs, int rstr, int roff,
                  float* Out, int nrows, int relu) {
    k_gemm128<<<(nrows + 15) / 16, 256, 0, stream>>>(
        A, W, bias, cs, sh, BNM, rs, rstr, roff, Out, nrows, relu);
  };

  const int EB  = (NE + NN + 255) / 256;                       // edge(+self) grid
  const int SB  = (int)(((long long)(NE + NN) * HD + 255) / 256);
  const int NFB = (NN * HD + 255) / 256;
  const int NB  = (NN + 255) / 256;

  // x = relu(bn_feat(x) @ conv_feat_W + b)
  gemm(x, P[6], P[7], P[4], P[5], nullptr, 0, 0, X, NN, 1);

  // 2 GAT layers
  for (int i = 0; i < 2; ++i) {
    const int b0 = 20 + i * 6;
    gemm(X, P[b0 + 2], nullptr, P[b0], P[b0 + 1], nullptr, 0, 0, T, NN, 0);  // h = bn(x)@W
    k_alpha<<<(NN * NH + 255) / 256, 256, 0, stream>>>(T, P[b0 + 3], P[b0 + 4], asrc, adst);
    hipMemsetAsync(mkey, 0, (size_t)NN * NH * 4, stream);
    hipMemsetAsync(ssum, 0, (size_t)NN * NH * 4, stream);
    k_edge_logit<<<EB, 256, 0, stream>>>(row, col, asrc, adst, att, mkey);
    k_edge_exp<<<EB, 256, 0, stream>>>(row, col, mkey, att, ssum);
    hipMemsetAsync(S, 0, (size_t)NN * HD * 4, stream);
    k_gat_scatter<<<SB, 256, 0, stream>>>(row, col, att, ssum, T, S);
    k_bias_relu<<<NFB, 256, 0, stream>>>(S, P[b0 + 5], X);
  }

  // edge / node attention
  k_edge_att<<<(NE + 255) / 256, 256, 0, stream>>>(row, col, X, P[8], P[9], eatt);
  k_node_att<<<NB, 256, 0, stream>>>(X, P[10], P[11], natt);

  // ctx branch GCN -> poolc
  hipMemsetAsync(deg, 0, (size_t)NN * 4, stream);
  k_deg<<<EB, 256, 0, stream>>>(row, eatt, 0, deg);
  k_dis<<<NB, 256, 0, stream>>>(deg);
  gemm(X, P[16], nullptr, P[12], P[13], natt, 2, 0, T, NN, 0);   // bn(natt0*x)@ctx_W
  hipMemsetAsync(S, 0, (size_t)NN * HD * 4, stream);
  k_gcn_scatter<<<SB, 256, 0, stream>>>(row, col, eatt, 0, deg, T, S);
  hipMemsetAsync(poolc, 0, (size_t)NG * HD * 4, stream);
  k_bias_relu_pool<<<NFB, 256, 0, stream>>>(S, P[17], batch, poolc);

  // obj branch GCN -> poolo
  hipMemsetAsync(deg, 0, (size_t)NN * 4, stream);
  k_deg<<<EB, 256, 0, stream>>>(row, eatt, 1, deg);
  k_dis<<<NB, 256, 0, stream>>>(deg);
  gemm(X, P[18], nullptr, P[14], P[15], natt, 2, 1, T, NN, 0);   // bn(natt1*x)@obj_W
  hipMemsetAsync(S, 0, (size_t)NN * HD * 4, stream);
  k_gcn_scatter<<<SB, 256, 0, stream>>>(row, col, eatt, 1, deg, T, S);
  hipMemsetAsync(poolo, 0, (size_t)NG * HD * 4, stream);
  k_bias_relu_pool<<<NFB, 256, 0, stream>>>(S, P[19], batch, poolo);

  // readouts
  float* out = (float*)d_out;
  // c
  gemm(poolc, P[34], P[35], P[32], P[33], nullptr, 0, 0, R, NG, 1);
  k_head<<<NG, 32, 0, stream>>>(R, P[36], P[37], BNM, P[38], P[39], out);
  // o
  gemm(poolo, P[42], P[43], P[40], P[41], nullptr, 0, 0, R, NG, 1);
  k_head<<<NG, 32, 0, stream>>>(R, P[44], P[45], BNM, P[46], P[47], out + NG * NC);
  // co : xc[perm] + xo
  k_combine<<<(NG * HD + 255) / 256, 256, 0, stream>>>(poolc, poolo, perm, pco);
  gemm(pco, P[50], P[51], P[48], P[49], nullptr, 0, 0, R, NG, 1);
  k_head<<<NG, 32, 0, stream>>>(R, P[52], P[53], BNM, P[54], P[55], out + 2 * NG * NC);
}
